// QMeasurement_48661979463911
// MI455X (gfx1250) — matrix operand order repaired and checked by reading, not compiled, not run on hardware
//
#include <hip/hip_runtime.h>
#include <hip/hip_bf16.h>

// Problem constants (reference: B=8192, DX=512, DY=4)
#define BATCH   8192
#define DXDIM   512
#define NCOL    16             // DY*DY (l,n) pairs -> WMMA N
#define KC      32             // WMMA K per instruction (bf16)
#define NKC     (DXDIM / KC)   // 16 k-chunks
#define NTILES  (BATCH / 16)   // 512 batch tiles of 16
#define WAVES_PER_WG 8
#define WG_THREADS   (WAVES_PER_WG * 32)
#define BGROUPS      (NTILES / WAVES_PER_WG)  // 64 batch groups
#define MPB     2              // m-values staged per LDS slab
#define MSEGS   4              // m-range split across blockIdx.y
#define PAIRS_PER_SEG (DXDIM / (MPB * MSEGS)) // 64

typedef __attribute__((ext_vector_type(16))) __bf16 v16bf;
typedef __attribute__((ext_vector_type(8)))  float  v8f;
typedef int v4i __attribute__((vector_size(16)));
typedef __attribute__((address_space(1))) v4i* gv4i_p;   // global AS pointer
typedef __attribute__((address_space(3))) v4i* lv4i_p;   // LDS AS pointer

// ws layout: [0,8MB) packed A (x bf16, WMMA A-layout)
//            [8MB,16MB) packed B (rho bf16, WMMA B-layout)
//            [16MB,18MB) f32 partial M tiles: [seg][tile][row][col]
#define XA_BYTES   ((size_t)NTILES * NKC * 32 * 32)   // 8 MB
#define RHO_OFF    XA_BYTES
#define RHO_BYTES  ((size_t)DXDIM * NKC * 32 * 32)    // 8 MB
#define PART_OFF   (RHO_OFF + RHO_BYTES)

#if defined(__has_builtin)
#if __has_builtin(__builtin_amdgcn_global_load_async_to_lds_b128)
#define HAVE_ASYNC_LDS 1
#endif
#endif

__device__ __forceinline__ __bf16 tobf(float f) { return (__bf16)f; }

// ---------------------------------------------------------------------------
// Pre-pass 1: pack x -> bf16 WMMA 16-bit A-matrix layout (ISA 7.12.2):
//   lanes 0-15 : row=lane,    K pairs {0..7,16..23}
//   lanes 16-31: row=lane-16, K pairs {8..15,24..31}
// ---------------------------------------------------------------------------
__global__ void pack_x_kernel(const float* __restrict__ x,
                              __bf16* __restrict__ xa) {
    int gid  = blockIdx.x * blockDim.x + threadIdx.x;   // 512*16*32 threads
    int lane = gid & 31;
    int kc   = (gid >> 5) & (NKC - 1);
    int tile = gid >> 9;

    int row   = lane & 15;
    int base0 = (lane < 16) ? 0 : 8;
    const float* xr = x + (size_t)(tile * 16 + row) * DXDIM + kc * KC;

    __bf16 h[16];
#pragma unroll
    for (int v = 0; v < 8; ++v) {
        int kk = (v < 4) ? (base0 + 2 * v) : (base0 + 16 + 2 * (v - 4));
        h[2 * v]     = tobf(xr[kk]);
        h[2 * v + 1] = tobf(xr[kk + 1]);
    }
    v16bf* dst = (v16bf*)(xa) + ((size_t)(tile * NKC + kc) * 32 + lane);
    *dst = *(const v16bf*)h;
}

// ---------------------------------------------------------------------------
// Pre-pass 2: pack rho (idx = k*8192 + l*2048 + m*4 + n) -> bf16 WMMA
// 16-bit B-matrix (32x16) layout per (m, kchunk):
//   lanes 0-15 : col=lane,    K pairs {0..7,16..23}
//   lanes 16-31: col=lane-16, K pairs {8..15,24..31}
// ---------------------------------------------------------------------------
__global__ void pack_rho_kernel(const float* __restrict__ rho,
                                __bf16* __restrict__ rb) {
    int gid  = blockIdx.x * blockDim.x + threadIdx.x;   // 512*16*32 threads
    int lane = gid & 31;
    int kc   = (gid >> 5) & (NKC - 1);
    int m    = gid >> 9;

    int col   = lane & 15;
    int kbase = kc * KC + ((lane < 16) ? 0 : 8);
    int l = col >> 2, n = col & 3;
    const float* rp = rho + (size_t)l * 2048 + (size_t)m * 4 + n;

    __bf16 h[16];
#pragma unroll
    for (int v = 0; v < 8; ++v) {
        int k0 = kbase + ((v < 4) ? 2 * v : 16 + 2 * (v - 4));
        h[2 * v]     = tobf(rp[(size_t)k0 * 8192]);
        h[2 * v + 1] = tobf(rp[(size_t)(k0 + 1) * 8192]);
    }
    v16bf* dst = (v16bf*)(rb) + ((size_t)(m * NKC + kc) * 32 + lane);
    *dst = *(const v16bf*)h;
}

// ---------------------------------------------------------------------------
// Staging helpers: move one 32 KB rho slab (2 m-slices) to LDS.
// ---------------------------------------------------------------------------
__device__ __forceinline__ void stage_rho_slab(const __bf16* rb, int pair,
                                               __bf16* dst, int tid) {
    const char* src = (const char*)rb + (size_t)pair * (MPB * NKC * 32 * 32);
#if HAVE_ASYNC_LDS
    char* dstc = (char*)dst;
#pragma unroll
    for (int i = 0; i < 8; ++i) {
        __builtin_amdgcn_global_load_async_to_lds_b128(
            (gv4i_p)(src + tid * 16 + i * 4096),
            (lv4i_p)(dstc + tid * 16 + i * 4096),
            0, 0);
    }
#else
    const v16bf* s = (const v16bf*)src;
    v16bf* d = (v16bf*)dst;
#pragma unroll
    for (int i = 0; i < 4; ++i)
        d[tid + i * 256] = s[tid + i * 256];
#endif
}

__device__ __forceinline__ void wait_async_stage() {
#if HAVE_ASYNC_LDS
#if __has_builtin(__builtin_amdgcn_s_wait_asynccnt)
    __builtin_amdgcn_s_wait_asynccnt(0);
#else
    asm volatile("s_wait_asynccnt 0x0" ::: "memory");
#endif
#endif
}

// ---------------------------------------------------------------------------
// Main kernel: grid (64 batch-groups, 4 m-segments) x 256 threads (8 waves).
// Wave w owns batch tile (bg*8+w); A resident in VGPRs; rho slabs double-
// buffered in LDS and shared by all 8 waves; two independent WMMA chains
// (one per staged m) per iteration. Partial tiles -> ws.
// ---------------------------------------------------------------------------
__global__ void __launch_bounds__(WG_THREADS)
qmeasure_wmma_kernel(const float* __restrict__ x,
                     const __bf16* __restrict__ xa,
                     const __bf16* __restrict__ rb,
                     float* __restrict__ pws) {
    __shared__ __align__(32) __bf16 sB[2][MPB * NKC * 32 * 16]; // 2 x 32 KB
    __shared__ float sxm[2][MPB][WAVES_PER_WG * 16];            // x[b,m] scales

    const int tid  = threadIdx.x;
    const int lane = tid & 31;
    const int wave = tid >> 5;
    const int bg   = blockIdx.x;
    const int seg  = blockIdx.y;
    const int batchtile = bg * WAVES_PER_WG + wave;
    const int p0 = seg * PAIRS_PER_SEG;

    // Resident A: 16 chunks of (16x32 bf16) = 128 VGPRs.
    v16bf a[NKC];
    const v16bf* ap = (const v16bf*)xa + (size_t)batchtile * NKC * 32 + lane;
#pragma unroll
    for (int kc = 0; kc < NKC; ++kc) a[kc] = ap[kc * 32];

    // Prologue: stage first slab into buffer 0.
    stage_rho_slab(rb, p0, &sB[0][0], tid);
    {
        int mm = p0 * MPB + (tid >> 7);
        sxm[0][tid >> 7][tid & 127] =
            x[(size_t)(bg * 128 + (tid & 127)) * DXDIM + mm];
    }
    wait_async_stage();
    __syncthreads();

    v8f acc = {};
    const int rowoff = (lane < 16) ? 0 : 8;

    for (int lp = 0; lp < PAIRS_PER_SEG; ++lp) {
        const int cur = lp & 1, nxt = cur ^ 1;
        // Stage next slab while computing current one.
        if (lp + 1 < PAIRS_PER_SEG) {
            stage_rho_slab(rb, p0 + lp + 1, &sB[nxt][0], tid);
            int mm = (p0 + lp + 1) * MPB + (tid >> 7);
            sxm[nxt][tid >> 7][tid & 127] =
                x[(size_t)(bg * 128 + (tid & 127)) * DXDIM + mm];
        }
        if (lp + 2 < PAIRS_PER_SEG)
            __builtin_prefetch((const char*)rb +
                (size_t)(p0 + lp + 2) * (MPB * NKC * 32 * 32) + tid * 128, 0, 0);

        // Two independent WMMA chains: m0 = 2*(p0+lp), m1 = m0+1.
        v8f t0 = {}, t1 = {};
        const v16bf* b0 = (const v16bf*)&sB[cur][0];
        const v16bf* b1 = (const v16bf*)&sB[cur][NKC * 32 * 16];
#pragma unroll
        for (int kc = 0; kc < NKC; ++kc) {
            v16bf bm0 = b0[kc * 32 + lane];
            v16bf bm1 = b1[kc * 32 + lane];
            t0 = __builtin_amdgcn_wmma_f32_16x16x32_bf16(
                     false, a[kc], false, bm0, (short)0, t0, false, false);
            t1 = __builtin_amdgcn_wmma_f32_16x16x32_bf16(
                     false, a[kc], false, bm1, (short)0, t1, false, false);
        }

        // acc[b,(l,n)] += x[b,m0]*T_m0 + x[b,m1]*T_m1 (row scale, per lane).
        const float* x0 = &sxm[cur][0][wave * 16 + rowoff];
        const float* x1 = &sxm[cur][1][wave * 16 + rowoff];
#pragma unroll
        for (int r = 0; r < 8; ++r) {
            acc[r] += x0[r] * t0[r];
            acc[r] += x1[r] * t1[r];
        }

        wait_async_stage();   // staged slab landed in LDS
        __syncthreads();      // all waves done reading cur; nxt visible
    }

    // Write partial tile (f32 C/D layout: VGPR r -> rows r / r+8).
    float* pdst = pws + ((size_t)(seg * NTILES + batchtile) * 256);
    const int col = lane & 15;
#pragma unroll
    for (int r = 0; r < 8; ++r)
        pdst[(r + rowoff) * 16 + col] = acc[r];
}

// ---------------------------------------------------------------------------
// Finalize: sum 4 m-segment partials, normalize by trace.
// One thread per batch row (8192 threads).
// ---------------------------------------------------------------------------
__global__ void finalize_kernel(const float* __restrict__ pws,
                                float* __restrict__ out) {
    int b = blockIdx.x * blockDim.x + threadIdx.x;   // 0..8191
    int tile = b >> 4, row = b & 15;

    float v[NCOL];
#pragma unroll
    for (int c = 0; c < NCOL; ++c) v[c] = 0.0f;
#pragma unroll
    for (int s = 0; s < MSEGS; ++s) {
        const float* p = pws + ((size_t)(s * NTILES + tile) * 256) + row * 16;
#pragma unroll
        for (int c = 0; c < NCOL; ++c) v[c] += p[c];
    }
    float tr  = v[0] + v[5] + v[10] + v[15];
    float inv = 1.0f / tr;
#pragma unroll
    for (int c = 0; c < NCOL; ++c) out[(size_t)b * NCOL + c] = v[c] * inv;
}

// ---------------------------------------------------------------------------
extern "C" void kernel_launch(void* const* d_in, const int* in_sizes, int n_in,
                              void* d_out, int out_size, void* d_ws, size_t ws_size,
                              hipStream_t stream) {
    const float* x   = (const float*)d_in[0];   // (8192, 512)
    const float* rho = (const float*)d_in[1];   // (512, 4, 512, 4)
    float* out = (float*)d_out;                 // (8192, 4, 4)

    __bf16* xa  = (__bf16*)d_ws;                          // 8 MB packed A
    __bf16* rb  = (__bf16*)((char*)d_ws + RHO_OFF);       // 8 MB packed B
    float*  pws = (float*)((char*)d_ws + PART_OFF);       // 2 MB partials

    pack_x_kernel<<<1024, 256, 0, stream>>>(x, xa);
    pack_rho_kernel<<<1024, 256, 0, stream>>>(rho, rb);

    dim3 grid(BGROUPS, MSEGS);   // 64 x 4 = 256 WGs, 2048 waves
    qmeasure_wmma_kernel<<<grid, WG_THREADS, 0, stream>>>(x, xa, rb, pws);

    finalize_kernel<<<BATCH / 256, 256, 0, stream>>>(pws, out);
}
